// Nexus_8375186227274
// MI455X (gfx1250) — matrix operand order repaired and checked by reading, not compiled, not run on hardware
//
#include <hip/hip_runtime.h>
#include <hip/hip_bf16.h>

// GPT-2 forward for MI455X (gfx1250, wave32, WMMA + async-to-LDS).
// Fast path: weights pre-converted to f16 [N][K]; activations f16; GEMM tiles
// double-buffered in LDS via GLOBAL_LOAD_ASYNC_TO_LDS_B128 (ASYNCcnt, in-order
// completion -> s_wait_asynccnt 5 overlaps next-tile loads with 8 WMMAs).
// f16 WMMA 16x16x32, fp32 accumulate; residual/softmax/LN stats fp32.
// Fallback (small ws): fp32-weight WMMA GEMM with on-the-fly conversion.

typedef __attribute__((ext_vector_type(16))) _Float16 v16h;
typedef __attribute__((ext_vector_type(8)))  _Float16 v8h;
typedef __attribute__((ext_vector_type(4)))  _Float16 v4h;
typedef __attribute__((ext_vector_type(8)))  float    v8f;

#define GPT_L   24
#define GPT_H   16
#define GPT_D   1024
#define GPT_V   50257
#define GPT_B   2
#define GPT_T   1024
#define GPT_M   (GPT_B * GPT_T)
#define GPT_HD  64

// GEMM tiling: 256 threads = 8 waves; C tile 256(M) x 64(N); wave tile 32x64.
#define BM 256
#define BN 64
#define BK 32
#define LDP 40    // fallback-path padded LDS stride (halves)

enum { EPI_NONE = 0, EPI_BIAS = 1, EPI_BIAS_RESID = 2, EPI_BIAS_GELU = 3 };

__device__ __forceinline__ float gelu_tanh(float x) {
    const float c0 = 0.7978845608028654f;
    const float c1 = 0.044715f;
    return 0.5f * x * (1.0f + tanhf(c0 * (x + c1 * x * x * x)));
}

// Async copy of 16 bytes global -> LDS (per lane). ASYNCcnt-tracked.
__device__ __forceinline__ void async_copy_b128(void* lds, const void* gptr) {
    unsigned lo = (unsigned)(uintptr_t)lds;            // flat addr[31:0] == LDS offset
    unsigned long long ga = (unsigned long long)(uintptr_t)gptr;
    asm volatile("global_load_async_to_lds_b128 %0, %1, off"
                 :: "v"(lo), "v"(ga) : "memory");
}
template <int CNT>
__device__ __forceinline__ void wait_async_le() {
#if __has_builtin(__builtin_amdgcn_s_wait_asynccnt)
    __builtin_amdgcn_s_wait_asynccnt(CNT);
#else
    asm volatile("s_wait_asynccnt %0" :: "i"(CNT) : "memory");
#endif
}

// ---------------------------------------------------------------------------
// Embedding: x[b,t,:] = wte[idx[b,t],:] + wpe[t,:]   (fp32)
// ---------------------------------------------------------------------------
__global__ void embed_kernel(const int* __restrict__ idx,
                             const float* __restrict__ wte,
                             const float* __restrict__ wpe,
                             float* __restrict__ x) {
    int row = blockIdx.x;
    int t   = row % GPT_T;
    int id  = idx[row];
    const float4* wrow = (const float4*)(wte + (size_t)id * GPT_D);
    const float4* prow = (const float4*)(wpe + (size_t)t  * GPT_D);
    float4* out = (float4*)(x + (size_t)row * GPT_D);
    for (int i = threadIdx.x; i < GPT_D / 4; i += blockDim.x) {
        float4 a = wrow[i], p = prow[i];
        out[i] = make_float4(a.x + p.x, a.y + p.y, a.z + p.z, a.w + p.w);
    }
}

// ---------------------------------------------------------------------------
// Weight conversion kernels (once per launch, fast path only)
// ---------------------------------------------------------------------------
// fp32 W[l][K][N] -> f16 Wt[l][N][K]  (32x32 LDS transpose tiles)
__global__ void convert_wT_kernel(const float* __restrict__ W,
                                  _Float16* __restrict__ Wt, int K, int N) {
    __shared__ float tile[32][33];
    int l  = blockIdx.z;
    const float*  Wl  = W  + (size_t)l * K * N;
    _Float16*     Wtl = Wt + (size_t)l * K * N;
    int nb = blockIdx.x * 32, kb = blockIdx.y * 32;
    int tx = threadIdx.x & 31, ty = threadIdx.x >> 5;      // 256 thr: ty 0..7
#pragma unroll
    for (int i = 0; i < 32; i += 8)
        tile[ty + i][tx] = Wl[(size_t)(kb + ty + i) * N + (nb + tx)];
    __syncthreads();
#pragma unroll
    for (int i = 0; i < 32; i += 8)
        Wtl[(size_t)(nb + ty + i) * K + (kb + tx)] = (_Float16)tile[tx][ty + i];
}

// fp32 rows [GPT_D] -> f16 rows (wte: already n-major)
__global__ void convert_rows_kernel(const float* __restrict__ in,
                                    _Float16* __restrict__ out) {
    size_t row = blockIdx.x;
    const float4* ip = (const float4*)(in + row * GPT_D);
    _Float16*     op = out + row * GPT_D;
    int i = threadIdx.x;                 // 256 threads * 4 = 1024
    float4 f = ip[i];
    v4h h; h[0] = (_Float16)f.x; h[1] = (_Float16)f.y;
    h[2] = (_Float16)f.z; h[3] = (_Float16)f.w;
    *(v4h*)(op + i * 4) = h;
}

// ---------------------------------------------------------------------------
// LayerNorm (fp32 stats), output fp32 or f16
// ---------------------------------------------------------------------------
template <typename OutT>
__global__ void ln_kernel(const float* __restrict__ x,
                          const float* __restrict__ w,
                          const float* __restrict__ b,
                          OutT* __restrict__ out) {
    __shared__ float red[256];
    int row = blockIdx.x;
    int tid = threadIdx.x;
    const float4* xr = (const float4*)(x + (size_t)row * GPT_D);

    float4 v = xr[tid];
    red[tid] = v.x + v.y + v.z + v.w; __syncthreads();
    for (int st = 128; st > 0; st >>= 1) {
        if (tid < st) red[tid] += red[tid + st];
        __syncthreads();
    }
    float mean = red[0] * (1.0f / GPT_D);
    __syncthreads();

    float dx = v.x - mean, dy = v.y - mean, dz = v.z - mean, dw = v.w - mean;
    red[tid] = dx * dx + dy * dy + dz * dz + dw * dw; __syncthreads();
    for (int st = 128; st > 0; st >>= 1) {
        if (tid < st) red[tid] += red[tid + st];
        __syncthreads();
    }
    float inv = rsqrtf(red[0] * (1.0f / GPT_D) + 1e-5f);

    const float4* w4 = (const float4*)w;
    const float4* b4 = (const float4*)b;
    float4 ww = w4[tid], bb = b4[tid];
    OutT* o = out + (size_t)row * GPT_D + tid * 4;
    o[0] = (OutT)(dx * inv * ww.x + bb.x);
    o[1] = (OutT)(dy * inv * ww.y + bb.y);
    o[2] = (OutT)(dz * inv * ww.z + bb.z);
    o[3] = (OutT)(dw * inv * ww.w + bb.w);
}

// ---------------------------------------------------------------------------
// FAST PATH GEMM: C[M,N] = A16[M,K] @ Wt16[N,K]^T (+bias fp32, +resid/gelu).
// Double-buffered async-to-LDS staging; 5 async ops per wave per stage
// (uniform: W row clamped for the N tail, garbage columns never stored).
// ---------------------------------------------------------------------------
template <typename OutT>
__global__ __launch_bounds__(256)
void gemm16_wmma_kernel(const _Float16* __restrict__ A,
                        const _Float16* __restrict__ Wt,
                        const float* __restrict__ bias,
                        const float* __restrict__ resid,
                        OutT* __restrict__ C,
                        int M, int N, int K, int epi) {
    __shared__ __align__(16) _Float16 As[2][BM][BK];   // 2 x 16 KB
    __shared__ __align__(16) _Float16 Ws[2][BN][BK];   // 2 x  4 KB

    const int tid  = threadIdx.x;
    const int wv   = tid >> 5;
    const int lane = tid & 31;
    const int m0   = blockIdx.x * BM;
    const int n0   = blockIdx.y * BN;

    const int mrow = lane & 15;
    const int kbA  = (lane >> 4) * 16;
    const int nloc = lane & 15;
    const int kbB  = (lane >> 4) * 16;
    const int rtop = (lane >> 4) * 8;

    v8f acc[2][4];
#pragma unroll
    for (int s = 0; s < 2; ++s)
#pragma unroll
        for (int t = 0; t < 4; ++t)
#pragma unroll
            for (int i = 0; i < 8; ++i) acc[s][t][i] = 0.0f;

    // per-thread staging units
    const int arow = tid >> 2, ag = tid & 3;     // A: 4 units (row, 8-half group)
    const int wn   = tid >> 2, wg = tid & 3;     // W: 1 unit
    const int gnW  = (n0 + wn < N) ? (n0 + wn) : (N - 1);   // clamped (uniform cnt)

    // stage: exactly 5 async b128 per thread (uniform per wave)
    auto stage = [&](int buf, int k0) {
#pragma unroll
        for (int i = 0; i < 4; ++i) {
            int row = arow + i * 64;
            async_copy_b128(&As[buf][row][ag * 8],
                            A + (size_t)(m0 + row) * K + k0 + ag * 8);
        }
        async_copy_b128(&Ws[buf][wn][wg * 8],
                        Wt + (size_t)gnW * K + k0 + wg * 8);
    };

    const int nkt = K / BK;
    stage(0, 0);

    for (int kt = 0; kt < nkt; ++kt) {
        const int cur = kt & 1;
        if (kt + 1 < nkt) {
            stage(cur ^ 1, (kt + 1) * BK);   // overlap next tile with compute
            wait_async_le<5>();              // oldest 5 (= current tile) landed
        } else {
            wait_async_le<0>();
        }
        __syncthreads();

        v16h af[2];
#pragma unroll
        for (int s = 0; s < 2; ++s)
#pragma unroll
            for (int i = 0; i < 16; ++i) {
                int kk = kbA + i;
                af[s][i] = As[cur][wv * 32 + s * 16 + mrow][kk];
            }
#pragma unroll
        for (int t = 0; t < 4; ++t) {
            v16h bf;
#pragma unroll
            for (int i = 0; i < 16; ++i)
                bf[i] = Ws[cur][t * 16 + nloc][kbB + i];
            acc[0][t] = __builtin_amdgcn_wmma_f32_16x16x32_f16(
                false, af[0], false, bf, (short)0, acc[0][t], false, false);
            acc[1][t] = __builtin_amdgcn_wmma_f32_16x16x32_f16(
                false, af[1], false, bf, (short)0, acc[1][t], false, false);
        }
        __syncthreads();   // all waves done reading buf[cur] before it's restaged
    }

    // --- epilogue ---
#pragma unroll
    for (int s = 0; s < 2; ++s) {
        const int mb = m0 + wv * 32 + s * 16 + rtop;
#pragma unroll
        for (int t = 0; t < 4; ++t) {
            int n = n0 + t * 16 + nloc;
            if (n >= N) continue;
            float bv = (epi != EPI_NONE) ? bias[n] : 0.0f;
#pragma unroll
            for (int r = 0; r < 8; ++r) {
                size_t o = (size_t)(mb + r) * N + n;
                float v = acc[s][t][r] + bv;
                if (epi == EPI_BIAS_GELU)       v = gelu_tanh(v);
                else if (epi == EPI_BIAS_RESID) v += resid[o];
                C[o] = (OutT)v;
            }
        }
    }
}

// ---------------------------------------------------------------------------
// FALLBACK GEMM (fp32 weights, on-the-fly f16 conversion).
// ---------------------------------------------------------------------------
__global__ __launch_bounds__(256)
void gemm_wmma_kernel(const float* __restrict__ A,
                      const float* __restrict__ W,
                      const float* __restrict__ bias,
                      const float* __restrict__ resid,
                      float* __restrict__ C,
                      int M, int N, int K, int ldw, int epi, int wtrans) {
    __shared__ __align__(16) _Float16 As[BM][LDP];
    __shared__ __align__(16) _Float16 Ws[BN][LDP];

    const int tid  = threadIdx.x;
    const int wv   = tid >> 5;
    const int lane = tid & 31;
    const int m0   = blockIdx.x * BM;
    const int n0   = blockIdx.y * BN;

    const int mrow = lane & 15;
    const int kbA  = (lane >> 4) * 16;
    const int nloc = lane & 15;
    const int kbB  = (lane >> 4) * 16;
    const int rtop = (lane >> 4) * 8;

    v8f acc[2][4];
#pragma unroll
    for (int s = 0; s < 2; ++s)
#pragma unroll
        for (int t = 0; t < 4; ++t)
#pragma unroll
            for (int i = 0; i < 8; ++i) acc[s][t][i] = 0.0f;

    for (int k0 = 0; k0 < K; k0 += BK) {
#pragma unroll
        for (int i = 0; i < (BM * BK) / (8 * 256); ++i) {
            int u   = tid + i * 256;
            int row = u >> 2;
            int g   = u & 3;
            const float* ap = A + (size_t)(m0 + row) * K + (k0 + g * 8);
            float4 f0 = *(const float4*)ap;
            float4 f1 = *(const float4*)(ap + 4);
            v8h hh;
            hh[0] = (_Float16)f0.x; hh[1] = (_Float16)f0.y;
            hh[2] = (_Float16)f0.z; hh[3] = (_Float16)f0.w;
            hh[4] = (_Float16)f1.x; hh[5] = (_Float16)f1.y;
            hh[6] = (_Float16)f1.z; hh[7] = (_Float16)f1.w;
            *(v8h*)&As[row][g * 8] = hh;
        }
        {
            int n  = tid >> 2;
            int g  = tid & 3;
            int ks = g * 8;
            v8h hh;
            if (wtrans) {
                int gn = n0 + n;
                if (gn < N) {
                    const float* wp = W + (size_t)gn * ldw + (k0 + ks);
                    float4 f0 = *(const float4*)wp;
                    float4 f1 = *(const float4*)(wp + 4);
                    hh[0] = (_Float16)f0.x; hh[1] = (_Float16)f0.y;
                    hh[2] = (_Float16)f0.z; hh[3] = (_Float16)f0.w;
                    hh[4] = (_Float16)f1.x; hh[5] = (_Float16)f1.y;
                    hh[6] = (_Float16)f1.z; hh[7] = (_Float16)f1.w;
                } else {
#pragma unroll
                    for (int j = 0; j < 8; ++j) hh[j] = (_Float16)0.0f;
                }
            } else {
                float f[8];
#pragma unroll
                for (int j = 0; j < 8; ++j)
                    f[j] = W[(size_t)(k0 + ks + j) * ldw + (n0 + n)];
#pragma unroll
                for (int j = 0; j < 8; ++j) hh[j] = (_Float16)f[j];
            }
            *(v8h*)&Ws[n][ks] = hh;
        }
        __syncthreads();

        v16h af[2];
#pragma unroll
        for (int s = 0; s < 2; ++s)
#pragma unroll
            for (int i = 0; i < 16; ++i) {
                int kk = kbA + i;
                af[s][i] = As[wv * 32 + s * 16 + mrow][kk];
            }
#pragma unroll
        for (int t = 0; t < 4; ++t) {
            v16h bf;
#pragma unroll
            for (int i = 0; i < 16; ++i)
                bf[i] = Ws[t * 16 + nloc][kbB + i];
            acc[0][t] = __builtin_amdgcn_wmma_f32_16x16x32_f16(
                false, af[0], false, bf, (short)0, acc[0][t], false, false);
            acc[1][t] = __builtin_amdgcn_wmma_f32_16x16x32_f16(
                false, af[1], false, bf, (short)0, acc[1][t], false, false);
        }
        __syncthreads();
    }

#pragma unroll
    for (int s = 0; s < 2; ++s) {
        const int mb = m0 + wv * 32 + s * 16 + rtop;
#pragma unroll
        for (int t = 0; t < 4; ++t) {
            int n = n0 + t * 16 + nloc;
            if (n >= N) continue;
            float bv = (epi != EPI_NONE) ? bias[n] : 0.0f;
#pragma unroll
            for (int r = 0; r < 8; ++r) {
                size_t o = (size_t)(mb + r) * N + n;
                float v = acc[s][t][r] + bv;
                if (epi == EPI_BIAS_GELU)       v = gelu_tanh(v);
                else if (epi == EPI_BIAS_RESID) v += resid[o];
                C[o] = v;
            }
        }
    }
}

// ---------------------------------------------------------------------------
// Causal attention, fp32 math; output fp32 or f16.
// ---------------------------------------------------------------------------
template <typename OutT>
__global__ void attn_kernel(const float* __restrict__ qkv,
                            OutT* __restrict__ y) {
    __shared__ float qs[GPT_HD];
    __shared__ float sc[GPT_T];
    __shared__ float red[128];

    int bhq = blockIdx.x;
    int q   = bhq & (GPT_T - 1);
    int bh  = bhq >> 10;
    int h   = bh & (GPT_H - 1);
    int b   = bh >> 4;
    int tid = threadIdx.x;

    const size_t rowQ = (size_t)(b * GPT_T + q) * (3 * GPT_D);
    if (tid < GPT_HD) qs[tid] = qkv[rowQ + h * GPT_HD + tid];
    __syncthreads();

    const int nk = q + 1;
    const float scale = 0.125f;

    float lmax = -1e30f;
    for (int k = tid; k < nk; k += 128) {
        const float4* kp = (const float4*)(qkv + (size_t)(b * GPT_T + k) * (3 * GPT_D)
                                           + GPT_D + h * GPT_HD);
        float d = 0.f;
#pragma unroll
        for (int c = 0; c < GPT_HD / 4; ++c) {
            float4 kv = kp[c];
            d += qs[c * 4 + 0] * kv.x + qs[c * 4 + 1] * kv.y
               + qs[c * 4 + 2] * kv.z + qs[c * 4 + 3] * kv.w;
        }
        d *= scale;
        sc[k] = d;
        lmax = fmaxf(lmax, d);
    }
    red[tid] = lmax; __syncthreads();
    for (int st = 64; st > 0; st >>= 1) {
        if (tid < st) red[tid] = fmaxf(red[tid], red[tid + st]);
        __syncthreads();
    }
    float mx = red[0];
    __syncthreads();

    float lsum = 0.f;
    for (int k = tid; k < nk; k += 128) {
        float e = __expf(sc[k] - mx);
        sc[k] = e;
        lsum += e;
    }
    red[tid] = lsum; __syncthreads();
    for (int st = 64; st > 0; st >>= 1) {
        if (tid < st) red[tid] += red[tid + st];
        __syncthreads();
    }
    float inv = 1.0f / red[0];
    __syncthreads();

    int c    = tid & (GPT_HD - 1);
    int part = tid >> 6;
    float accv = 0.f;
    for (int k = part; k < nk; k += 2) {
        accv += sc[k] * qkv[(size_t)(b * GPT_T + k) * (3 * GPT_D) + 2 * GPT_D + h * GPT_HD + c];
    }
    red[tid] = accv; __syncthreads();
    if (tid < GPT_HD) {
        y[(size_t)(b * GPT_T + q) * GPT_D + h * GPT_HD + tid] =
            (OutT)((red[tid] + red[tid + 64]) * inv);
    }
}

// ---------------------------------------------------------------------------
// Host-side orchestration (graph-capture safe: launches only).
// ---------------------------------------------------------------------------
extern "C" void kernel_launch(void* const* d_in, const int* in_sizes, int n_in,
                              void* d_out, int out_size, void* d_ws, size_t ws_size,
                              hipStream_t stream) {
    const int*   idx     = (const int*)  d_in[0];
    const float* wte     = (const float*)d_in[1];
    const float* wpe     = (const float*)d_in[2];
    const float* ln1_w   = (const float*)d_in[3];
    const float* ln1_b   = (const float*)d_in[4];
    const float* attn_w  = (const float*)d_in[5];
    const float* attn_b  = (const float*)d_in[6];
    const float* aproj_w = (const float*)d_in[7];
    const float* aproj_b = (const float*)d_in[8];
    const float* ln2_w   = (const float*)d_in[9];
    const float* ln2_b   = (const float*)d_in[10];
    const float* fc_w    = (const float*)d_in[11];
    const float* fc_b    = (const float*)d_in[12];
    const float* proj_w  = (const float*)d_in[13];
    const float* proj_b  = (const float*)d_in[14];
    const float* lnf_w   = (const float*)d_in[15];
    const float* lnf_b   = (const float*)d_in[16];
    float* logits = (float*)d_out;

    const dim3 blk256(256);
    const size_t MD  = (size_t)GPT_M * GPT_D;

    // ---- fast-path workspace budget (bytes) ----
    const size_t sz_x     = MD * 4;
    const size_t sz_qkv   = MD * 3 * 4;
    const size_t sz_h16   = MD * 2;
    const size_t sz_y16   = MD * 2;
    const size_t sz_h2_16 = MD * 4 * 2;
    const size_t sz_wa    = (size_t)GPT_L * GPT_D * 3 * GPT_D * 2;
    const size_t sz_wp    = (size_t)GPT_L * GPT_D * GPT_D * 2;
    const size_t sz_wf    = (size_t)GPT_L * GPT_D * 4 * GPT_D * 2;
    const size_t sz_wo    = (size_t)GPT_L * 4 * GPT_D * GPT_D * 2;
    const size_t sz_wte16 = (size_t)GPT_V * GPT_D * 2;
    const size_t need16 = sz_x + sz_qkv + sz_h16 + sz_y16 + sz_h2_16
                        + sz_wa + sz_wp + sz_wf + sz_wo + sz_wte16;

    if (ws_size >= need16) {
        // =========================== FAST PATH ===========================
        char* p = (char*)d_ws;
        float*     x     = (float*)p;       p += sz_x;
        float*     qkv   = (float*)p;       p += sz_qkv;
        _Float16*  h16   = (_Float16*)p;    p += sz_h16;
        _Float16*  y16   = (_Float16*)p;    p += sz_y16;
        _Float16*  h2_16 = (_Float16*)p;    p += sz_h2_16;
        _Float16*  wa16  = (_Float16*)p;    p += sz_wa;
        _Float16*  wp16  = (_Float16*)p;    p += sz_wp;
        _Float16*  wf16  = (_Float16*)p;    p += sz_wf;
        _Float16*  wo16  = (_Float16*)p;    p += sz_wo;
        _Float16*  wte16 = (_Float16*)p;    p += sz_wte16;

        convert_wT_kernel<<<dim3(3 * GPT_D / 32, GPT_D / 32, GPT_L), blk256, 0, stream>>>(
            attn_w, wa16, GPT_D, 3 * GPT_D);
        convert_wT_kernel<<<dim3(GPT_D / 32, GPT_D / 32, GPT_L), blk256, 0, stream>>>(
            aproj_w, wp16, GPT_D, GPT_D);
        convert_wT_kernel<<<dim3(4 * GPT_D / 32, GPT_D / 32, GPT_L), blk256, 0, stream>>>(
            fc_w, wf16, GPT_D, 4 * GPT_D);
        convert_wT_kernel<<<dim3(GPT_D / 32, 4 * GPT_D / 32, GPT_L), blk256, 0, stream>>>(
            proj_w, wo16, 4 * GPT_D, GPT_D);
        convert_rows_kernel<<<GPT_V, blk256, 0, stream>>>(wte, wte16);

        embed_kernel<<<GPT_M, blk256, 0, stream>>>(idx, wte, wpe, x);

        for (int l = 0; l < GPT_L; ++l) {
            ln_kernel<_Float16><<<GPT_M, blk256, 0, stream>>>(
                x, ln1_w + (size_t)l * GPT_D, ln1_b + (size_t)l * GPT_D, h16);
            gemm16_wmma_kernel<float><<<dim3(GPT_M / BM, (3 * GPT_D) / BN), blk256, 0, stream>>>(
                h16, wa16 + (size_t)l * GPT_D * 3 * GPT_D,
                attn_b + (size_t)l * 3 * GPT_D, nullptr, qkv,
                GPT_M, 3 * GPT_D, GPT_D, EPI_BIAS);
            attn_kernel<_Float16><<<GPT_B * GPT_H * GPT_T, dim3(128), 0, stream>>>(qkv, y16);
            gemm16_wmma_kernel<float><<<dim3(GPT_M / BM, GPT_D / BN), blk256, 0, stream>>>(
                y16, wp16 + (size_t)l * GPT_D * GPT_D,
                aproj_b + (size_t)l * GPT_D, x, x,
                GPT_M, GPT_D, GPT_D, EPI_BIAS_RESID);
            ln_kernel<_Float16><<<GPT_M, blk256, 0, stream>>>(
                x, ln2_w + (size_t)l * GPT_D, ln2_b + (size_t)l * GPT_D, h16);
            gemm16_wmma_kernel<_Float16><<<dim3(GPT_M / BM, (4 * GPT_D) / BN), blk256, 0, stream>>>(
                h16, wf16 + (size_t)l * GPT_D * 4 * GPT_D,
                fc_b + (size_t)l * 4 * GPT_D, nullptr, h2_16,
                GPT_M, 4 * GPT_D, GPT_D, EPI_BIAS_GELU);
            gemm16_wmma_kernel<float><<<dim3(GPT_M / BM, GPT_D / BN), blk256, 0, stream>>>(
                h2_16, wo16 + (size_t)l * 4 * GPT_D * GPT_D,
                proj_b + (size_t)l * GPT_D, x, x,
                GPT_M, GPT_D, 4 * GPT_D, EPI_BIAS_RESID);
        }

        ln_kernel<_Float16><<<GPT_M, blk256, 0, stream>>>(x, lnf_w, lnf_b, h16);
        gemm16_wmma_kernel<float><<<dim3(GPT_M / BM, (GPT_V + BN - 1) / BN), blk256, 0, stream>>>(
            h16, wte16, nullptr, nullptr, logits, GPT_M, GPT_V, GPT_D, EPI_NONE);
    } else {
        // ========================== FALLBACK PATH =========================
        float* ws  = (float*)d_ws;
        float* x   = ws;
        float* h   = x   + MD;
        float* qkv = h   + MD;
        float* y   = qkv + MD * 3;
        float* h2  = y   + MD;

        embed_kernel<<<GPT_M, blk256, 0, stream>>>(idx, wte, wpe, x);
        for (int l = 0; l < GPT_L; ++l) {
            const float* aw = attn_w + (size_t)l * GPT_D * 3 * GPT_D;
            const float* pw = aproj_w + (size_t)l * GPT_D * GPT_D;
            const float* fw = fc_w + (size_t)l * GPT_D * 4 * GPT_D;
            const float* ow = proj_w + (size_t)l * 4 * GPT_D * GPT_D;

            ln_kernel<float><<<GPT_M, blk256, 0, stream>>>(
                x, ln1_w + (size_t)l * GPT_D, ln1_b + (size_t)l * GPT_D, h);
            gemm_wmma_kernel<<<dim3(GPT_M / BM, (3 * GPT_D) / BN), blk256, 0, stream>>>(
                h, aw, attn_b + (size_t)l * 3 * GPT_D, nullptr, qkv,
                GPT_M, 3 * GPT_D, GPT_D, 3 * GPT_D, EPI_BIAS, 0);
            attn_kernel<float><<<GPT_B * GPT_H * GPT_T, dim3(128), 0, stream>>>(qkv, y);
            gemm_wmma_kernel<<<dim3(GPT_M / BM, GPT_D / BN), blk256, 0, stream>>>(
                y, pw, aproj_b + (size_t)l * GPT_D, x, x,
                GPT_M, GPT_D, GPT_D, GPT_D, EPI_BIAS_RESID, 0);
            ln_kernel<float><<<GPT_M, blk256, 0, stream>>>(
                x, ln2_w + (size_t)l * GPT_D, ln2_b + (size_t)l * GPT_D, h);
            gemm_wmma_kernel<<<dim3(GPT_M / BM, (4 * GPT_D) / BN), blk256, 0, stream>>>(
                h, fw, fc_b + (size_t)l * 4 * GPT_D, nullptr, h2,
                GPT_M, 4 * GPT_D, GPT_D, 4 * GPT_D, EPI_BIAS_GELU, 0);
            gemm_wmma_kernel<<<dim3(GPT_M / BM, GPT_D / BN), blk256, 0, stream>>>(
                h2, ow, proj_b + (size_t)l * GPT_D, x, x,
                GPT_M, GPT_D, 4 * GPT_D, GPT_D, EPI_BIAS_RESID, 0);
        }
        ln_kernel<float><<<GPT_M, blk256, 0, stream>>>(x, lnf_w, lnf_b, h);
        gemm_wmma_kernel<<<dim3(GPT_M / BM, (GPT_V + BN - 1) / BN), blk256, 0, stream>>>(
            h, wte, nullptr, nullptr, logits, GPT_M, GPT_V, GPT_D, GPT_D, EPI_NONE, 1);
    }
}
